// ExponentialConcordanceLoss_89962384982541
// MI455X (gfx1250) — compile-verified, matrix operand order checked by reading, not run on hardware
//
#include <hip/hip_runtime.h>

typedef __attribute__((ext_vector_type(16))) _Float16 v16h;
typedef __attribute__((ext_vector_type(8)))  float    v8f;

#define BCHUNK 2048
#define BLOCK  256
#define ROWS_PER_WG 128   // 8 waves * 16 rows

union HFrag { v16h h; unsigned int u[8]; };

// loss = sum_{a,b} [dur_b < dur_a][evt_b==1] * (exp(p_a)*w_a) * exp(-p_b)
// cnt  = sum_{a,b} [dur_b < dur_a][evt_b==1]
__global__ __launch_bounds__(BLOCK) void concordance_main(
    const float* __restrict__ preds,
    const float* __restrict__ targets,   // interleaved [dur, evt]
    float* __restrict__ partials, int N)
{
  __shared__ float        sdur[BCHUNK];
  __shared__ unsigned int sg[BCHUNK];          // {Eh:f16hi, Gh:f16lo}
  __shared__ float redL[BLOCK / 32];
  __shared__ float redC[BLOCK / 32];

  const int tid  = threadIdx.x;
  const int lane = tid & 31;
  const int wave = tid >> 5;
  const int m    = lane & 15;
  const int half = lane >> 4;
  const int wgRow0 = blockIdx.x * ROWS_PER_WG;
  const int myRow  = wgRow0 + wave * 16 + m;

  float dur_a = -1.0f;                          // durations >= 0 -> mask always false
  if (myRow < N) dur_a = targets[2 * myRow];

  HFrag ones;                                   // B operand: all 1.0h (layout-free)
  #pragma unroll
  for (int i = 0; i < 8; ++i) ones.u[i] = 0x3C003C00u;

  v8f accG = {};                                // per-row  sum mask*Gh
  v8f accE = {};                                // per-row  pair count

  const int nchunks = (N + BCHUNK - 1) / BCHUNK;
  for (int c = 0; c < nchunks; ++c) {
    const int cb = c * BCHUNK;

    // --- stage b-side chunk into LDS: duration + packed {evt-flag, f16(exp(-p))} ---
    for (int idx = tid; idx < BCHUNK; idx += BLOCK) {
      const int gi = cb + idx;
      float d = 3.0e38f;                        // +big: never < dur_a
      unsigned int pk = 0u;
      if (gi < N) {
        const float2 t = *(const float2*)(targets + 2 * gi);
        d = t.x;
        if (t.y == 1.0f) {
          const float G = __expf(-preds[gi]);   // in [e^-6, e^6]: safe in f16
          union { _Float16 f; unsigned short s; } cv;
          cv.f = (_Float16)G;                   // v_cvt_f16_f32, RTN
          pk = 0x3C000000u | (unsigned int)cv.s;
        }
      }
      sdur[idx] = d;
      sg[idx]   = pk;
    }
    if (c + 1 < nchunks) {                      // global_prefetch_b8 of next chunk
      const int nb = cb + BCHUNK;
      __builtin_prefetch(targets + 2 * nb + tid * 16, 0, 0);
      __builtin_prefetch(preds + nb + tid * 8, 0, 0);
    }
    __syncthreads();

    // --- sweep chunk in K=32 slabs; one 16x16x32 WMMA pair per slab ---
    #pragma unroll 2
    for (int it = 0; it < BCHUNK / 32; ++it) {
      const int b0 = it * 32;
      HFrag fG, fE;
      #pragma unroll
      for (int r = 0; r < 2; ++r) {
        // A-frag 16x32 f16 layout: run r covers K = 16*r + 8*half + [0,8)
        const float*        dp = sdur + b0 + r * 16 + 8 * half;
        const unsigned int* gp = sg   + b0 + r * 16 + 8 * half;
        const float4 d0 = *(const float4*)dp;
        const float4 d1 = *(const float4*)(dp + 4);
        const uint4  g0 = *(const uint4*)gp;
        const uint4  g1 = *(const uint4*)(gp + 4);
        const float dd[8] = {d0.x, d0.y, d0.z, d0.w, d1.x, d1.y, d1.z, d1.w};
        const unsigned int gg[8] = {g0.x, g0.y, g0.z, g0.w, g1.x, g1.y, g1.z, g1.w};
        #pragma unroll
        for (int q = 0; q < 4; ++q) {
          const unsigned int s0 = (dd[2 * q]     < dur_a) ? gg[2 * q]     : 0u;
          const unsigned int s1 = (dd[2 * q + 1] < dur_a) ? gg[2 * q + 1] : 0u;
          // one v_perm_b32 each instead of and+lshl_or pairs
          fG.u[r * 4 + q] = __builtin_amdgcn_perm(s1, s0, 0x05040100u); // lo halves
          fE.u[r * 4 + q] = __builtin_amdgcn_perm(s1, s0, 0x07060302u); // hi halves
        }
      }
      accG = __builtin_amdgcn_wmma_f32_16x16x32_f16(
                 false, fG.h, false, ones.h, (short)0, accG, false, false);
      accE = __builtin_amdgcn_wmma_f32_16x16x32_f16(
                 false, fE.h, false, ones.h, (short)0, accE, false, false);
    }
    __syncthreads();
  }

  // --- fold in A-side factor exp(p_a)*w_a; D row = r + 8*half ---
  float lossP = 0.0f, cntP = 0.0f;
  const int rbase = wgRow0 + wave * 16 + 8 * half;
  #pragma unroll
  for (int r = 0; r < 8; ++r) {
    const int row = rbase + r;
    float aval = 0.0f;
    if (row < N) {
      const float evt = targets[2 * row + 1];
      aval = __expf(preds[row]) * ((evt == 1.0f) ? 1.0f : 0.5f);
    }
    lossP += aval * accG[r];
    cntP  += accE[r];
  }
  // each row value is replicated across the 16 identical D columns -> /16 below
  #pragma unroll
  for (int off = 16; off > 0; off >>= 1) {
    lossP += __shfl_xor(lossP, off, 32);
    cntP  += __shfl_xor(cntP,  off, 32);
  }
  if (lane == 0) { redL[wave] = lossP; redC[wave] = cntP; }
  __syncthreads();
  if (tid == 0) {
    float L = 0.0f, C = 0.0f;
    #pragma unroll
    for (int w = 0; w < BLOCK / 32; ++w) { L += redL[w]; C += redC[w]; }
    partials[2 * blockIdx.x]     = L * (1.0f / 16.0f);
    partials[2 * blockIdx.x + 1] = C * (1.0f / 16.0f);
  }
}

__global__ void concordance_finalize(const float* __restrict__ partials,
                                     float* __restrict__ out, int nwg)
{
  if (threadIdx.x == 0 && blockIdx.x == 0) {
    float L = 0.0f, C = 0.0f;
    for (int i = 0; i < nwg; ++i) { L += partials[2 * i]; C += partials[2 * i + 1]; }
    out[0] = (C > 0.0f) ? (L / C) : 0.0f;
  }
}

extern "C" void kernel_launch(void* const* d_in, const int* in_sizes, int n_in,
                              void* d_out, int out_size, void* d_ws, size_t ws_size,
                              hipStream_t stream) {
  (void)n_in; (void)out_size; (void)ws_size;
  const float* preds   = (const float*)d_in[0];
  const float* targets = (const float*)d_in[1];
  float* out = (float*)d_out;
  const int N   = in_sizes[0];
  const int nwg = (N + ROWS_PER_WG - 1) / ROWS_PER_WG;
  float* partials = (float*)d_ws;               // nwg*2 floats, rewritten every call

  concordance_main<<<nwg, BLOCK, 0, stream>>>(preds, targets, partials, N);
  concordance_finalize<<<1, 32, 0, stream>>>(partials, out, nwg);
}
